// SelfAttention_14705968021937
// MI455X (gfx1250) — matrix operand order repaired and checked by reading, not compiled, not run on hardware
//
#include <hip/hip_runtime.h>

// ---------------- problem constants ----------------
constexpr int Bb = 2;
constexpr int Tt = 2048;
constexpr int Cc = 1024;
constexpr int Hh = 16;
constexpr int Dd = 64;
constexpr int Mrows = Bb * Tt;   // 4096
constexpr int K1 = Cc;           // 1024
constexpr int N1 = 3 * Cc;       // 3072

typedef unsigned short u16;
typedef unsigned int u32;
typedef __attribute__((ext_vector_type(16))) __bf16 v16bf;
typedef __attribute__((ext_vector_type(8)))  unsigned short v8u16;
typedef __attribute__((ext_vector_type(8)))  float v8f;
typedef int v4i __attribute__((__vector_size__(4 * sizeof(int))));

union F16x16 {
    v16bf bf;
    v8u16 h[2];
    u16   u[16];
};

__device__ __forceinline__ u16 f2bf(float f) {
    u32 u = __float_as_uint(f);
    u32 r = u + 0x7FFFu + ((u >> 16) & 1u);   // round-to-nearest-even
    return (u16)(r >> 16);
}

__device__ __forceinline__ v8u16 ld8(const u16* p) {
    return *reinterpret_cast<const v8u16*>(p);
}

__device__ __forceinline__ v8f wmma_bf16(const F16x16& a, const F16x16& b, v8f c) {
    return __builtin_amdgcn_wmma_f32_16x16x32_bf16(
        /*neg_a=*/false, a.bf, /*neg_b=*/false, b.bf,
        /*c_mod=*/(short)0, c, /*reuse_a=*/false, /*reuse_b=*/false);
}

#define NEG_INF (-__builtin_inff())

// ---------------- CDNA5 async global->LDS (ASYNCcnt) with safe fallback ----------
#if defined(__has_builtin)
#if __has_builtin(__builtin_amdgcn_global_load_async_to_lds_b128) && \
    __has_builtin(__builtin_amdgcn_s_wait_asynccnt)
#define USE_ASYNC_LDS 1
#endif
#endif

__device__ __forceinline__ void async_cp16(const u16* gsrc, u16* ldst) {
#ifdef USE_ASYNC_LDS
    __builtin_amdgcn_global_load_async_to_lds_b128(
        (v4i*)(void*)gsrc, (v4i*)(void*)ldst, 0, 0);
#else
    *reinterpret_cast<uint4*>(ldst) = *reinterpret_cast<const uint4*>(gsrc);
#endif
}

__device__ __forceinline__ void wait_async() {
#ifdef USE_ASYNC_LDS
    __builtin_amdgcn_s_wait_asynccnt(0);
#endif
}

// ---------------- fp32 -> bf16 elementwise (x) ----------------
__global__ void cvt_x_kernel(const float* __restrict__ x, u16* __restrict__ xb, int n4) {
    int i = blockIdx.x * blockDim.x + threadIdx.x;
    if (i >= n4) return;
    float4 f = reinterpret_cast<const float4*>(x)[i];
    u32 lo = (u32)f2bf(f.x) | ((u32)f2bf(f.y) << 16);
    u32 hi = (u32)f2bf(f.z) | ((u32)f2bf(f.w) << 16);
    reinterpret_cast<uint2*>(xb)[i] = make_uint2(lo, hi);
}

// ---------------- fp32 [K][N] -> bf16 [N][K] (weights, transposed) ----------------
__global__ void cvt_transpose_kernel(const float* __restrict__ W, u16* __restrict__ Wt,
                                     int K, int N) {
    __shared__ float tile[32][33];
    int tx = threadIdx.x, ty = threadIdx.y;
    int k = blockIdx.y * 32 + ty;
    int n = blockIdx.x * 32 + tx;
    tile[ty][tx] = W[(size_t)k * N + n];
    __syncthreads();
    int no = blockIdx.x * 32 + ty;
    int ko = blockIdx.y * 32 + tx;
    Wt[(size_t)no * K + ko] = f2bf(tile[tx][ty]);
}

// ---------------- bf16 GEMM: C[M,N] = A[M,K] * Bt[N,K]^T + bias ----------------
// 128x128 block tile, 8 waves, each wave 32x64; K-chunk 64; double-buffered
// async global->LDS staging.
// mode 0: scatter into Q/K/V bf16 buffers [B,H,T,D]   (QKV projection)
// mode 1: write fp32 rows to outf [M,N]               (output projection)
__global__ __launch_bounds__(256) void gemm_bf16_kernel(
    const u16* __restrict__ A, const u16* __restrict__ Bt, const float* __restrict__ bias,
    int M, int N, int K, int mode,
    u16* __restrict__ qbuf, u16* __restrict__ kbuf, u16* __restrict__ vbuf,
    float* __restrict__ outf)
{
    __shared__ __align__(16) u16 As[2][128][64];   // 2 x 16 KB
    __shared__ __align__(16) u16 Bs[2][128][64];   // 2 x 16 KB

    const int tid   = threadIdx.x;
    const int mBase = blockIdx.y * 128;
    const int nBase = blockIdx.x * 128;
    const int w     = tid >> 5;
    const int lane  = tid & 31;
    const int m16   = lane & 15;
    const int hi    = lane >> 4;
    const int waveM = w >> 1;      // 0..3, 32 rows each
    const int waveN = w & 1;       // 0..1, 64 cols each

    v8f acc[2][4] = {};

    // cooperative tile staging: 128x64 bf16 for A and B, 32 elems/thread each
    auto load_tiles = [&](int buf, int kb0) {
        int row = tid >> 1, seg = tid & 1;
        const u16* asrc = A  + (size_t)(mBase + row) * K + kb0 + seg * 32;
        const u16* bsrc = Bt + (size_t)(nBase + row) * K + kb0 + seg * 32;
        u16* adst = &As[buf][row][seg * 32];
        u16* bdst = &Bs[buf][row][seg * 32];
#pragma unroll
        for (int i = 0; i < 4; i++) {
            async_cp16(asrc + i * 8, adst + i * 8);
            async_cp16(bsrc + i * 8, bdst + i * 8);
        }
    };

    load_tiles(0, 0);
    wait_async();
    __syncthreads();

    const int nTiles = K / 64;
    for (int t = 0; t < nTiles; t++) {
        const int cur = t & 1;
        if (t + 1 < nTiles) load_tiles(cur ^ 1, (t + 1) * 64);   // prefetch next tile (async)

#pragma unroll
        for (int kk = 0; kk < 64; kk += 32) {
            F16x16 a0, a1;
            const u16* ap0 = &As[cur][waveM * 32 + m16][kk];
            const u16* ap1 = &As[cur][waveM * 32 + 16 + m16][kk];
            a0.h[0] = ld8(ap0 + hi * 8);
            a0.h[1] = ld8(ap0 + 16 + hi * 8);
            a1.h[0] = ld8(ap1 + hi * 8);
            a1.h[1] = ld8(ap1 + 16 + hi * 8);
#pragma unroll
            for (int nt = 0; nt < 4; nt++) {
                F16x16 bfrag;
                const u16* bp = &Bs[cur][waveN * 64 + nt * 16 + m16][kk + hi * 8];
                bfrag.h[0] = ld8(bp);
                bfrag.h[1] = ld8(bp + 16);
                acc[0][nt] = wmma_bf16(a0, bfrag, acc[0][nt]);
                acc[1][nt] = wmma_bf16(a1, bfrag, acc[1][nt]);
            }
        }
        wait_async();
        __syncthreads();
    }

    // epilogue
#pragma unroll
    for (int part = 0; part < 2; part++) {
#pragma unroll
        for (int nt = 0; nt < 4; nt++) {
#pragma unroll
            for (int r = 0; r < 8; r++) {
                int ml = waveM * 32 + part * 16 + hi * 8 + r;
                int nl = waveN * 64 + nt * 16 + m16;
                size_t mg = (size_t)(mBase + ml);
                size_t ng = (size_t)(nBase + nl);
                float v = acc[part][nt][r] + bias[ng];
                if (mode == 1) {
                    outf[mg * (size_t)N + ng] = v;
                } else {
                    int sect = (int)(ng >> 10);          // 0=q 1=k 2=v
                    int c    = (int)(ng & 1023);
                    int hh   = c >> 6;
                    int dd   = c & 63;
                    int bb   = (int)(mg >> 11);
                    int tt   = (int)(mg & 2047);
                    size_t idx = (((size_t)bb * Hh + hh) * Tt + tt) * Dd + dd;
                    u16 bv = f2bf(v);
                    u16* dst = (sect == 0) ? qbuf : (sect == 1) ? kbuf : vbuf;
                    dst[idx] = bv;
                }
            }
        }
    }
}

// ---------------- causal flash attention ----------------
// grid: (T/64, H, B), block: 128 (4 waves); wave w owns q-rows [qblk*64+w*16, +16)
__global__ __launch_bounds__(128) void attn_kernel(
    const u16* __restrict__ Qb, const u16* __restrict__ Kb, const u16* __restrict__ Vb,
    u16* __restrict__ Aout)
{
    __shared__ __align__(16) u16 Klds[32][64];      // [key][d]        4 KB
    __shared__ __align__(16) u16 Vlds[64][40];      // [d][key] (pad)  5 KB
    __shared__ __align__(16) u16 Plds[4][16][32];   // per-wave P      4 KB

    const int tid  = threadIdx.x;
    const int w    = tid >> 5;
    const int lane = tid & 31;
    const int m16  = lane & 15;
    const int hi   = lane >> 4;
    const int qblk = blockIdx.x;
    const int h    = blockIdx.y;
    const int b    = blockIdx.z;
    const int qbase = qblk * 64 + w * 16;
    const size_t bh = (size_t)b * Hh + h;

    const u16* Qp = Qb + bh * Tt * Dd;
    const u16* Kp = Kb + bh * Tt * Dd;
    const u16* Vp = Vb + bh * Tt * Dd;

    // Q fragments (A-layout): lane holds row m16, d in {hi*8..}+{16+hi*8..} per 32-chunk
    F16x16 a0, a1;
    {
        const u16* qrow = Qp + (size_t)(qbase + m16) * Dd;
        a0.h[0] = ld8(qrow + hi * 8);
        a0.h[1] = ld8(qrow + 16 + hi * 8);
        a1.h[0] = ld8(qrow + 32 + hi * 8);
        a1.h[1] = ld8(qrow + 48 + hi * 8);
    }

    v8f O[4] = {};
    float mr[8], lr[8];
#pragma unroll
    for (int r = 0; r < 8; r++) { mr[r] = NEG_INF; lr[r] = 0.0f; }

    const int nKB = qblk * 2 + 2;  // key blocks of 32 covering [0, qblk*64+64)
    for (int jb = 0; jb < nKB; jb++) {
        const int kbase = jb * 32;
        // cooperative load: K via async global->LDS, V transposed via VGPRs
        {
            int row = tid >> 2, seg = tid & 3;   // 32 rows x 4 segs of 16
            const u16* ks = Kp + (size_t)(kbase + row) * Dd + seg * 16;
            async_cp16(ks,     &Klds[row][seg * 16]);
            async_cp16(ks + 8, &Klds[row][seg * 16 + 8]);
            const u16* vs = Vp + (size_t)(kbase + row) * Dd + seg * 16;
            union { uint4 q[2]; u16 s[16]; } tmp;
            tmp.q[0] = *reinterpret_cast<const uint4*>(vs);
            tmp.q[1] = *reinterpret_cast<const uint4*>(vs + 8);
#pragma unroll
            for (int i = 0; i < 16; i++) Vlds[seg * 16 + i][row] = tmp.s[i];
        }
        wait_async();
        __syncthreads();

        if (kbase <= qbase + 15) {   // wave-uniform causal skip
            v8f s0 = {}, s1 = {};
            F16x16 bk;
            // S tile 0: keys kbase + [0,16)
            bk.h[0] = ld8(&Klds[m16][hi * 8]);      bk.h[1] = ld8(&Klds[m16][hi * 8 + 16]);
            s0 = wmma_bf16(a0, bk, s0);
            bk.h[0] = ld8(&Klds[m16][32 + hi * 8]); bk.h[1] = ld8(&Klds[m16][32 + hi * 8 + 16]);
            s0 = wmma_bf16(a1, bk, s0);
            // S tile 1: keys kbase + [16,32)
            bk.h[0] = ld8(&Klds[16 + m16][hi * 8]);      bk.h[1] = ld8(&Klds[16 + m16][hi * 8 + 16]);
            s1 = wmma_bf16(a0, bk, s1);
            bk.h[0] = ld8(&Klds[16 + m16][32 + hi * 8]); bk.h[1] = ld8(&Klds[16 + m16][32 + hi * 8 + 16]);
            s1 = wmma_bf16(a1, bk, s1);

            const float scl = 0.125f;   // 1/sqrt(64)
            float alpha[8];
#pragma unroll
            for (int r = 0; r < 8; r++) {
                int q = qbase + hi * 8 + r;
                float v0 = s0[r] * scl;
                float v1 = s1[r] * scl;
                if (kbase + m16 > q)      v0 = NEG_INF;
                if (kbase + 16 + m16 > q) v1 = NEG_INF;
                // row max across 16 lanes (each half-wave group holds one row set)
                float rm = fmaxf(v0, v1);
                rm = fmaxf(rm, __shfl_xor(rm, 1, 32));
                rm = fmaxf(rm, __shfl_xor(rm, 2, 32));
                rm = fmaxf(rm, __shfl_xor(rm, 4, 32));
                rm = fmaxf(rm, __shfl_xor(rm, 8, 32));
                float mn = fmaxf(mr[r], rm);
                float al = __expf(mr[r] - mn);
                mr[r] = mn;
                v0 = __expf(v0 - mn);
                v1 = __expf(v1 - mn);
                float rs = v0 + v1;
                rs += __shfl_xor(rs, 1, 32);
                rs += __shfl_xor(rs, 2, 32);
                rs += __shfl_xor(rs, 4, 32);
                rs += __shfl_xor(rs, 8, 32);
                lr[r] = lr[r] * al + rs;
                alpha[r] = al;
                // stash P (bf16) in C-layout for re-layout into an A-fragment
                Plds[w][hi * 8 + r][m16]      = f2bf(v0);
                Plds[w][hi * 8 + r][16 + m16] = f2bf(v1);
            }
#pragma unroll
            for (int nt = 0; nt < 4; nt++)
#pragma unroll
                for (int r = 0; r < 8; r++) O[nt][r] *= alpha[r];

            // P as A-fragment (same-wave LDS ops are in-order; no barrier needed)
            F16x16 pa;
            pa.h[0] = ld8(&Plds[w][m16][hi * 8]);
            pa.h[1] = ld8(&Plds[w][m16][16 + hi * 8]);
#pragma unroll
            for (int nt = 0; nt < 4; nt++) {
                F16x16 vf;
                const u16* vp2 = &Vlds[nt * 16 + m16][hi * 8];
                vf.h[0] = ld8(vp2);
                vf.h[1] = ld8(vp2 + 16);
                O[nt] = wmma_bf16(pa, vf, O[nt]);
            }
        }
        __syncthreads();
    }

    // normalize and write attention output as bf16 rows [B*T, C] for out-proj GEMM
#pragma unroll
    for (int nt = 0; nt < 4; nt++) {
#pragma unroll
        for (int r = 0; r < 8; r++) {
            int t = qbase + hi * 8 + r;
            int d = nt * 16 + m16;
            float v = O[nt][r] / lr[r];
            Aout[((size_t)(b * Tt + t)) * Cc + h * Dd + d] = f2bf(v);
        }
    }
}

// ---------------- launch ----------------
extern "C" void kernel_launch(void* const* d_in, const int* in_sizes, int n_in,
                              void* d_out, int out_size, void* d_ws, size_t ws_size,
                              hipStream_t stream) {
    const float* x     = (const float*)d_in[0];
    const float* Wqkv  = (const float*)d_in[1];
    const float* bqkv  = (const float*)d_in[2];
    const float* Wproj = (const float*)d_in[3];
    const float* bproj = (const float*)d_in[4];
    float* out = (float*)d_out;

    char* ws = (char*)d_ws;
    size_t off = 0;
    auto take = [&](size_t bytes) -> void* {
        void* p = ws + off;
        off += (bytes + 255) & ~(size_t)255;
        return p;
    };
    u16* xb     = (u16*)take((size_t)Mrows * K1 * 2);        // 8 MB  bf16 x
    u16* Wqkvt  = (u16*)take((size_t)N1 * K1 * 2);           // 6 MB  bf16 W_qkv^T
    u16* Wprojt = (u16*)take((size_t)Cc * Cc * 2);           // 2 MB  bf16 W_proj^T
    u16* Qbuf   = (u16*)take((size_t)Bb * Hh * Tt * Dd * 2); // 8 MB
    u16* Kbuf   = (u16*)take((size_t)Bb * Hh * Tt * Dd * 2); // 8 MB
    u16* Vbuf   = (u16*)take((size_t)Bb * Hh * Tt * Dd * 2); // 8 MB
    u16* Aout   = (u16*)take((size_t)Mrows * Cc * 2);        // 8 MB  attention out (bf16)

    // 1. convert x to bf16
    int n4 = Mrows * K1 / 4;
    cvt_x_kernel<<<dim3((n4 + 255) / 256), dim3(256), 0, stream>>>(x, xb, n4);
    // 2. convert + transpose weights
    cvt_transpose_kernel<<<dim3(N1 / 32, K1 / 32), dim3(32, 32), 0, stream>>>(Wqkv, Wqkvt, K1, N1);
    cvt_transpose_kernel<<<dim3(Cc / 32, Cc / 32), dim3(32, 32), 0, stream>>>(Wproj, Wprojt, Cc, Cc);
    // 3. QKV projection (WMMA GEMM), scatter to Q/K/V [B,H,T,D] bf16
    gemm_bf16_kernel<<<dim3(N1 / 128, Mrows / 128), dim3(256), 0, stream>>>(
        xb, Wqkvt, bqkv, Mrows, N1, K1, /*mode=*/0, Qbuf, Kbuf, Vbuf, nullptr);
    // 4. causal flash attention (WMMA)
    attn_kernel<<<dim3(Tt / 64, Hh, Bb), dim3(128), 0, stream>>>(Qbuf, Kbuf, Vbuf, Aout);
    // 5. output projection (WMMA GEMM) -> fp32 d_out
    gemm_bf16_kernel<<<dim3(Cc / 128, Mrows / 128), dim3(256), 0, stream>>>(
        Aout, Wprojt, bproj, Mrows, Cc, K1, /*mode=*/1, nullptr, nullptr, nullptr, out);
}
